// ViT_MoMBlock_10179072491670
// MI455X (gfx1250) — matrix operand order repaired and checked by reading, not compiled, not run on hardware
//
#include <hip/hip_runtime.h>
#include <hip/hip_bf16.h>
#include <math.h>

// ---------------------------------------------------------------------------
// ViT MoM block for MI455X (gfx1250, wave32, WMMA f32_16x16x32_f16)
// B=64, S=196, D=768, E=4, K=2, TH=196, MH=3072
// S/TH padded to 224 (7 k-steps of 32), N side of mixer padded to 208 (13 tiles)
// MLP GEMMs register-blocked: 2 M-tiles x 4 N-tiles per wave (8 WMMAs / k-step)
// ---------------------------------------------------------------------------

typedef __attribute__((ext_vector_type(16))) _Float16 v16h;
typedef __attribute__((ext_vector_type(8)))  _Float16 v8h;
typedef __attribute__((ext_vector_type(8)))  float    v8f;

#define B_   64
#define S_   196
#define D_   768
#define E_   4
#define TH_  196
#define MH_  3072
#define SP_  224   // padded K for mixer (7*32)
#define NT13 13    // N tiles for mixer (13*16 = 208 >= 196)

__device__ __forceinline__ v8f zero8() {
  v8f z = {0.f,0.f,0.f,0.f,0.f,0.f,0.f,0.f};
  return z;
}

// A fragment (16x32, 16-bit): lane l holds row M=l&15; half-group h=l>>4 gives
// K chunks [8h, 8h+8) and [16+8h, 16+8h+8)  -> two aligned 16B loads.
__device__ __forceinline__ v16h load_a_frag(const _Float16* Ab, int lda,
                                            int kbase, int ml, int hh) {
  const _Float16* p = Ab + (long)ml * lda + kbase + hh * 8;
  v8h c0 = *(const v8h*)p;
  v8h c1 = *(const v8h*)(p + 16);
  v16h r;
#pragma unroll
  for (int i = 0; i < 8; ++i) { r[i] = c0[i]; r[i + 8] = c1[i]; }
  return r;
}

// B fragment: pre-packed fragment-major tile (512 halves); lane reads its 16
// contiguous halves (32B aligned) -> one v16h load.
__device__ __forceinline__ v16h load_b_frag(const _Float16* tile, int lane) {
  return *(const v16h*)(tile + lane * 16);
}

__device__ __forceinline__ v8f wmma_f16(v16h a, v16h b, v8f c) {
  return __builtin_amdgcn_wmma_f32_16x16x32_f16(false, a, false, b,
                                                (short)0, c, false, false);
}

__device__ __forceinline__ float gelu_exact(float v) {
  return 0.5f * v * (1.0f + erff(v * 0.70710678118654752f));
}

// ---------------------------------------------------------------------------
// Pack a row-major f32 [K x N] B-matrix into WMMA fragment-major f16 tiles.
// Tile (kt, nt) is 32x16; within a tile: lane = h*16 + n%16, half s maps to
// k%32 = 8h+s (s<8) or 16+8h+(s-8) (s>=8). Tile index = kt*NT + nt, 512 halves each. Zero-pads OOB.
// ---------------------------------------------------------------------------
__global__ void pack_b_kernel(const float* __restrict__ src,
                              _Float16* __restrict__ dst,
                              int K, int N, int KT, int NT,
                              long srcStride, long dstStride) {
  const int e = blockIdx.y;
  const long total = (long)KT * NT * 512;
  long idx = (long)blockIdx.x * blockDim.x + threadIdx.x;
  if (idx >= total) return;
  int tile = (int)(idx >> 9);
  int inT  = (int)(idx & 511);
  int lane = inT >> 4, half = inT & 15;
  int kt = tile / NT, nt = tile % NT;
  int kin = (half < 8) ? (lane >> 4) * 8 + half : 16 + (lane >> 4) * 8 + (half - 8);
  int n = nt * 16 + (lane & 15);
  int k = kt * 32 + kin;
  float v = (k < K && n < N) ? src[(long)e * srcStride + (long)k * N + n] : 0.0f;
  dst[(long)e * dstStride + idx] = (_Float16)v;
}

// ---------------------------------------------------------------------------
// LN1: per (b,s) token LayerNorm over D; writes transposed padded f16 matrix
// normedT[b][d][s] (width SP_=224, cols 196..223 zeroed).
// ---------------------------------------------------------------------------
__global__ void __launch_bounds__(256) ln1_kernel(const float* __restrict__ x,
                                                  const float* __restrict__ g,
                                                  const float* __restrict__ bt,
                                                  _Float16* __restrict__ normedT) {
  const int blk = blockIdx.x;
  const int b = blk / S_, s = blk % S_;
  const int tid = threadIdx.x;
  const float* row = x + (long)blk * D_;
  float v0 = row[tid], v1 = row[tid + 256], v2 = row[tid + 512];
  __shared__ float s1[256], s2[256];
  s1[tid] = v0 + v1 + v2;
  s2[tid] = v0 * v0 + v1 * v1 + v2 * v2;
  __syncthreads();
  for (int st = 128; st > 0; st >>= 1) {
    if (tid < st) { s1[tid] += s1[tid + st]; s2[tid] += s2[tid + st]; }
    __syncthreads();
  }
  const float mu = s1[0] * (1.0f / D_);
  const float var = s2[0] * (1.0f / D_) - mu * mu;
  const float rstd = rsqrtf(var + 1e-6f);
  float vv[3] = {v0, v1, v2};
#pragma unroll
  for (int j = 0; j < 3; ++j) {
    int d = tid + j * 256;
    float val = (vv[j] - mu) * rstd * g[d] + bt[d];
    normedT[((long)b * D_ + d) * SP_ + s] = (_Float16)val;
  }
  if (s < SP_ - S_) {  // zero pad columns 196..223 once
#pragma unroll
    for (int j = 0; j < 3; ++j) {
      int d = tid + j * 256;
      normedT[((long)b * D_ + d) * SP_ + S_ + s] = (_Float16)0.0f;
    }
  }
}

// ---------------------------------------------------------------------------
// Router: per-batch mean over tokens, logits = mean @ Wg + bg, softmax, top-2
// (ties -> lower index, matching jax.lax.top_k), renormalize.
// ---------------------------------------------------------------------------
__global__ void __launch_bounds__(256) router_kernel(
    const _Float16* __restrict__ normedT, const float* __restrict__ Wg,
    const float* __restrict__ bg, float* __restrict__ combine,
    float* __restrict__ probs) {
  const int b = blockIdx.x, tid = threadIdx.x;
  float acc[E_] = {0.f, 0.f, 0.f, 0.f};
  for (int d = tid; d < D_; d += 256) {
    const _Float16* r = normedT + ((long)b * D_ + d) * SP_;
    float sum = 0.f;
    for (int s = 0; s < S_; ++s) sum += (float)r[s];
    float mean = sum * (1.0f / S_);
#pragma unroll
    for (int e = 0; e < E_; ++e) acc[e] += mean * Wg[d * E_ + e];
  }
  __shared__ float sred[256];
  __shared__ float logit[E_];
  for (int e = 0; e < E_; ++e) {
    sred[tid] = acc[e];
    __syncthreads();
    for (int st = 128; st > 0; st >>= 1) {
      if (tid < st) sred[tid] += sred[tid + st];
      __syncthreads();
    }
    if (tid == 0) logit[e] = sred[0];
    __syncthreads();
  }
  if (tid == 0) {
    float lg[E_], mx = -1e30f;
    for (int e = 0; e < E_; ++e) { lg[e] = logit[e] + bg[e]; mx = fmaxf(mx, lg[e]); }
    float p[E_], se = 0.f;
    for (int e = 0; e < E_; ++e) { p[e] = expf(lg[e] - mx); se += p[e]; }
    for (int e = 0; e < E_; ++e) { p[e] /= se; probs[b * E_ + e] = p[e]; }
    int i1 = 0;
    for (int e = 1; e < E_; ++e) if (p[e] > p[i1]) i1 = e;
    int i2 = (i1 == 0) ? 1 : 0;
    for (int e = 0; e < E_; ++e) if (e != i1 && p[e] > p[i2]) i2 = e;
    float s2 = p[i1] + p[i2];
    float cmb[E_] = {0.f, 0.f, 0.f, 0.f};
    cmb[i1] = p[i1] / s2;
    cmb[i2] = p[i2] / s2;
    for (int e = 0; e < E_; ++e) combine[b * E_ + e] = cmb[e];
  }
}

// ---------------------------------------------------------------------------
// Fused token mixer: per (rb, b) a workgroup owns 32 D-rows.
// For each selected expert e: H = gelu(A @ W1[e] + b1[e]) staged in LDS
// (32 x 224 f16, 14KB), then O = H @ W2[e] + b2[e], weighted by combine and
// accumulated in registers. Finally x1 = x + sum_e c_e * O.  No atomics.
// All WMMA sites are wave/block-uniform -> EXEC all-ones.
// ---------------------------------------------------------------------------
__global__ void __launch_bounds__(256) mixer_kernel(
    const _Float16* __restrict__ normedT, const _Float16* __restrict__ W1p,
    const _Float16* __restrict__ W2p, const float* __restrict__ b1,
    const float* __restrict__ b2, const float* __restrict__ combine,
    const float* __restrict__ x, float* __restrict__ x1) {
  const int rb = blockIdx.x;   // 0..23  (D row-block of 32)
  const int b  = blockIdx.y;   // 0..63
  const int tid = threadIdx.x;
  const int wave = tid >> 5, lane = tid & 31;
  const int ml = lane & 15, hh = lane >> 4;
  __shared__ __align__(16) _Float16 Hs[32 * SP_];

  // zero pad cols 196..223 (never touched by phase-1 stores)
  for (int i = tid; i < 32 * (SP_ - S_); i += 256) {
    int r = i / (SP_ - S_), c = S_ + i % (SP_ - S_);
    Hs[r * SP_ + c] = (_Float16)0.0f;
  }
  v8f acc[4];
#pragma unroll
  for (int j = 0; j < 4; ++j) acc[j] = zero8();
  __syncthreads();

  for (int e = 0; e < E_; ++e) {
    const float c = combine[b * E_ + e];
    if (c == 0.0f) continue;  // block-uniform
    // ---- phase 1: H tiles (2 M-tiles x 13 N-tiles = 26 tiles over 8 waves)
    for (int t = wave; t < 2 * NT13; t += 8) {
      const int mt = t & 1, nt = t >> 1;
      v8f h = zero8();
      const _Float16* Abase = normedT + ((long)b * D_ + rb * 32 + mt * 16) * SP_;
#pragma unroll
      for (int kt = 0; kt < 7; ++kt) {
        v16h a  = load_a_frag(Abase, SP_, kt * 32, ml, hh);
        v16h bb = load_b_frag(W1p + ((long)e * 7 * NT13 + kt * NT13 + nt) * 512, lane);
        h = wmma_f16(a, bb, h);
      }
      const int n = nt * 16 + ml;
      if (n < TH_) {
        const float bias = b1[e * TH_ + n];
        const int mrow = mt * 16 + 8 * hh;
#pragma unroll
        for (int i = 0; i < 8; ++i) {
          float v = gelu_exact(h[i] + bias);
          Hs[(mrow + i) * SP_ + n] = (_Float16)v;
        }
      }
    }
    __syncthreads();
    // ---- phase 2: O = H @ W2[e] + b2[e], scaled by c into register acc
    for (int j = 0; j < 4; ++j) {
      const int t = wave + 8 * j;
      if (t >= 2 * NT13) break;  // wave-uniform
      const int mt = t & 1, nt = t >> 1;
      v8f o = zero8();
#pragma unroll
      for (int kt = 0; kt < 7; ++kt) {
        v16h a  = load_a_frag(&Hs[(mt * 16) * SP_], SP_, kt * 32, ml, hh);
        v16h bb = load_b_frag(W2p + ((long)e * 7 * NT13 + kt * NT13 + nt) * 512, lane);
        o = wmma_f16(a, bb, o);
      }
      const int n = nt * 16 + ml;
      const float bias = (n < S_) ? b2[e * S_ + n] : 0.0f;
#pragma unroll
      for (int i = 0; i < 8; ++i) acc[j][i] += c * (o[i] + bias);
    }
    __syncthreads();  // before Hs is rewritten for next expert
  }

  // ---- residual write: x1[b][s][d] = x + sum_e c_e * O   (each elem once)
  for (int j = 0; j < 4; ++j) {
    const int t = wave + 8 * j;
    if (t >= 2 * NT13) break;
    const int mt = t & 1, nt = t >> 1;
    const int s = nt * 16 + ml;
    if (s < S_) {
      const int d0 = rb * 32 + mt * 16 + 8 * hh;
      const long base = ((long)b * S_ + s) * D_ + d0;
#pragma unroll
      for (int i = 0; i < 8; ++i) x1[base + i] = x[base + i] + acc[j][i];
    }
  }
}

// ---------------------------------------------------------------------------
// LN2: per-row LayerNorm of x1 -> n2 (f16 row-major [12544 x 768])
// ---------------------------------------------------------------------------
__global__ void __launch_bounds__(256) ln2_kernel(const float* __restrict__ x1,
                                                  const float* __restrict__ g,
                                                  const float* __restrict__ bt,
                                                  _Float16* __restrict__ n2) {
  const int r = blockIdx.x;
  const int tid = threadIdx.x;
  const float* row = x1 + (long)r * D_;
  float v0 = row[tid], v1 = row[tid + 256], v2 = row[tid + 512];
  __shared__ float s1[256], s2[256];
  s1[tid] = v0 + v1 + v2;
  s2[tid] = v0 * v0 + v1 * v1 + v2 * v2;
  __syncthreads();
  for (int st = 128; st > 0; st >>= 1) {
    if (tid < st) { s1[tid] += s1[tid + st]; s2[tid] += s2[tid + st]; }
    __syncthreads();
  }
  const float mu = s1[0] * (1.0f / D_);
  const float var = s2[0] * (1.0f / D_) - mu * mu;
  const float rstd = rsqrtf(var + 1e-6f);
  float vv[3] = {v0, v1, v2};
#pragma unroll
  for (int j = 0; j < 3; ++j) {
    int d = tid + j * 256;
    n2[(long)r * D_ + d] = (_Float16)((vv[j] - mu) * rstd * g[d] + bt[d]);
  }
}

// ---------------------------------------------------------------------------
// MLP fc1: Hm = gelu(n2 [12544x768] @ Wm1 [768x3072] + bm1), f16 out.
// Register-blocked: each wave computes 2 M-tiles x 4 N-tiles (32x64);
// 8 waves arranged 2x4 -> block covers 64 M x 256 N. Grid (12, 196). K=24.
// Per k-step: 2 A frags + 4 B frags feed 8 WMMAs.
// ---------------------------------------------------------------------------
__global__ void __launch_bounds__(256) mlp_fc1_kernel(
    const _Float16* __restrict__ A, const _Float16* __restrict__ Bp,
    const float* __restrict__ bias, _Float16* __restrict__ H) {
  const int tid = threadIdx.x, wave = tid >> 5, lane = tid & 31;
  const int ml = lane & 15, hh = lane >> 4;
  const int mw = wave & 1;                     // 0..1
  const int nw = wave >> 1;                    // 0..3
  const int m0tile = blockIdx.y * 4 + mw * 2;  // 2 tiles: m0tile, m0tile+1 (<784)
  const int n0tile = blockIdx.x * 16 + nw * 4; // 4 tiles (<192)
  const _Float16* A0 = A + (long)m0tile * 16 * D_;
  const _Float16* A1 = A0 + 16 * D_;
  v8f acc[2][4];
#pragma unroll
  for (int mm = 0; mm < 2; ++mm)
#pragma unroll
    for (int nn = 0; nn < 4; ++nn) acc[mm][nn] = zero8();
#pragma unroll 2
  for (int kt = 0; kt < 24; ++kt) {
    v16h a0 = load_a_frag(A0, D_, kt * 32, ml, hh);
    v16h a1 = load_a_frag(A1, D_, kt * 32, ml, hh);
#pragma unroll
    for (int nn = 0; nn < 4; ++nn) {
      v16h bb = load_b_frag(Bp + ((long)kt * 192 + n0tile + nn) * 512, lane);
      acc[0][nn] = wmma_f16(a0, bb, acc[0][nn]);
      acc[1][nn] = wmma_f16(a1, bb, acc[1][nn]);
    }
  }
#pragma unroll
  for (int mm = 0; mm < 2; ++mm) {
    const int m0 = (m0tile + mm) * 16 + 8 * hh;
#pragma unroll
    for (int nn = 0; nn < 4; ++nn) {
      const int n = (n0tile + nn) * 16 + ml;
      const float bs = bias[n];
#pragma unroll
      for (int i = 0; i < 8; ++i) {
        float v = gelu_exact(acc[mm][nn][i] + bs);
        H[(long)(m0 + i) * MH_ + n] = (_Float16)v;
      }
    }
  }
}

// ---------------------------------------------------------------------------
// MLP fc2: out = x1 + Hm [12544x3072] @ Wm2 [3072x768] + bm2, f32 out.
// Same register blocking. Grid (3, 196). K=96.
// ---------------------------------------------------------------------------
__global__ void __launch_bounds__(256) mlp_fc2_kernel(
    const _Float16* __restrict__ A, const _Float16* __restrict__ Bp,
    const float* __restrict__ bias, const float* __restrict__ res,
    float* __restrict__ out) {
  const int tid = threadIdx.x, wave = tid >> 5, lane = tid & 31;
  const int ml = lane & 15, hh = lane >> 4;
  const int mw = wave & 1;
  const int nw = wave >> 1;
  const int m0tile = blockIdx.y * 4 + mw * 2;  // < 784
  const int n0tile = blockIdx.x * 16 + nw * 4; // < 48
  const _Float16* A0 = A + (long)m0tile * 16 * MH_;
  const _Float16* A1 = A0 + 16 * MH_;
  v8f acc[2][4];
#pragma unroll
  for (int mm = 0; mm < 2; ++mm)
#pragma unroll
    for (int nn = 0; nn < 4; ++nn) acc[mm][nn] = zero8();
#pragma unroll 2
  for (int kt = 0; kt < 96; ++kt) {
    v16h a0 = load_a_frag(A0, MH_, kt * 32, ml, hh);
    v16h a1 = load_a_frag(A1, MH_, kt * 32, ml, hh);
#pragma unroll
    for (int nn = 0; nn < 4; ++nn) {
      v16h bb = load_b_frag(Bp + ((long)kt * 48 + n0tile + nn) * 512, lane);
      acc[0][nn] = wmma_f16(a0, bb, acc[0][nn]);
      acc[1][nn] = wmma_f16(a1, bb, acc[1][nn]);
    }
  }
#pragma unroll
  for (int mm = 0; mm < 2; ++mm) {
    const int m0 = (m0tile + mm) * 16 + 8 * hh;
#pragma unroll
    for (int nn = 0; nn < 4; ++nn) {
      const int n = (n0tile + nn) * 16 + ml;
      const float bs = bias[n];
#pragma unroll
      for (int i = 0; i < 8; ++i) {
        const long idx = (long)(m0 + i) * D_ + n;
        out[idx] = res[idx] + acc[mm][nn][i] + bs;
      }
    }
  }
}

// ---------------------------------------------------------------------------
// Aux loss: E * sum_e mean_b(mask) * mean_b(probs); single scalar.
// ---------------------------------------------------------------------------
__global__ void aux_kernel(const float* __restrict__ probs,
                           const float* __restrict__ combine,
                           float* __restrict__ outp) {
  __shared__ float sm[E_], sp[E_];
  const int t = threadIdx.x;  // t = b*4 + e, 256 = B_*E_
  if (t < E_) { sm[t] = 0.f; sp[t] = 0.f; }
  __syncthreads();
  const int e = t & 3;
  atomicAdd(&sm[e], combine[t] > 0.0f ? 1.0f : 0.0f);
  atomicAdd(&sp[e], probs[t]);
  __syncthreads();
  if (t == 0) {
    float a = 0.f;
    for (int k = 0; k < E_; ++k) a += (sm[k] / B_) * (sp[k] / B_);
    *outp = (float)E_ * a;
  }
}

// ---------------------------------------------------------------------------
extern "C" void kernel_launch(void* const* d_in, const int* in_sizes, int n_in,
                              void* d_out, int out_size, void* d_ws, size_t ws_size,
                              hipStream_t stream) {
  const float* x   = (const float*)d_in[0];
  const float* g1  = (const float*)d_in[1];
  const float* be1 = (const float*)d_in[2];
  const float* Wg  = (const float*)d_in[3];
  const float* bg  = (const float*)d_in[4];
  const float* W1  = (const float*)d_in[5];
  const float* b1  = (const float*)d_in[6];
  const float* W2  = (const float*)d_in[7];
  const float* b2  = (const float*)d_in[8];
  const float* g2  = (const float*)d_in[9];
  const float* be2 = (const float*)d_in[10];
  const float* Wm1 = (const float*)d_in[11];
  const float* bm1 = (const float*)d_in[12];
  const float* Wm2 = (const float*)d_in[13];
  const float* bm2 = (const float*)d_in[14];
  float* out = (float*)d_out;

  char* base = (char*)d_ws;
  size_t off = 0;
  auto take = [&](size_t bytes) -> char* {
    off = (off + 255) & ~(size_t)255;
    char* p = base + off;
    off += bytes;
    return p;
  };
  float*    combine = (float*)take(B_ * E_ * sizeof(float));
  float*    probs   = (float*)take(B_ * E_ * sizeof(float));
  _Float16* normedT = (_Float16*)take((size_t)B_ * D_ * SP_ * 2);
  float*    x1      = (float*)take((size_t)B_ * S_ * D_ * 4);
  _Float16* n2      = (_Float16*)take((size_t)B_ * S_ * D_ * 2);
  _Float16* Hm      = (_Float16*)take((size_t)B_ * S_ * MH_ * 2);
  _Float16* W1p     = (_Float16*)take((size_t)E_ * 7 * NT13 * 512 * 2);
  _Float16* W2p     = (_Float16*)take((size_t)E_ * 7 * NT13 * 512 * 2);
  _Float16* Wm1p    = (_Float16*)take((size_t)24 * 192 * 512 * 2);
  _Float16* Wm2p    = (_Float16*)take((size_t)96 * 48 * 512 * 2);

  // pack weights into WMMA fragment-major tiles
  {
    long tot = 7L * NT13 * 512;
    dim3 g((unsigned)((tot + 255) / 256), E_);
    pack_b_kernel<<<g, 256, 0, stream>>>(W1, W1p, S_, TH_, 7, NT13,
                                         (long)S_ * TH_, tot);
    pack_b_kernel<<<g, 256, 0, stream>>>(W2, W2p, TH_, S_, 7, NT13,
                                         (long)TH_ * S_, tot);
  }
  {
    long tot1 = 24L * 192 * 512;
    pack_b_kernel<<<dim3((unsigned)((tot1 + 255) / 256), 1), 256, 0, stream>>>(
        Wm1, Wm1p, D_, MH_, 24, 192, 0, 0);
    long tot2 = 96L * 48 * 512;
    pack_b_kernel<<<dim3((unsigned)((tot2 + 255) / 256), 1), 256, 0, stream>>>(
        Wm2, Wm2p, MH_, D_, 96, 48, 0, 0);
  }

  ln1_kernel<<<B_ * S_, 256, 0, stream>>>(x, g1, be1, normedT);
  router_kernel<<<B_, 256, 0, stream>>>(normedT, Wg, bg, combine, probs);
  mixer_kernel<<<dim3(D_ / 32, B_), 256, 0, stream>>>(normedT, W1p, W2p, b1, b2,
                                                      combine, x, x1);
  ln2_kernel<<<B_ * S_, 256, 0, stream>>>(x1, g2, be2, n2);
  mlp_fc1_kernel<<<dim3(MH_ / 256, (B_ * S_) / 64), 256, 0, stream>>>(n2, Wm1p,
                                                                      bm1, Hm);
  mlp_fc2_kernel<<<dim3(D_ / 256, (B_ * S_) / 64), 256, 0, stream>>>(Hm, Wm2p,
                                                                     bm2, x1, out);
  aux_kernel<<<1, 256, 0, stream>>>(probs, combine, out + (size_t)B_ * S_ * D_);
}
